// ANI2xt_67602785239290
// MI455X (gfx1250) — compile-verified, matrix operand order checked by reading, not run on hardware
//
#include <hip/hip_runtime.h>
#include <hip/hip_bf16.h>
#include <cstdint>
#include <math.h>

typedef __attribute__((ext_vector_type(16))) _Float16 v16h;
typedef __attribute__((ext_vector_type(8)))  _Float16 v8h;
typedef __attribute__((ext_vector_type(8)))  float    v8f;

#define NSPEC   7
#define NATOM   32
#define NMOL    64
#define AEV_DIM 1008
#define AEV_PAD 1024
#define RCR_F   5.2f
#define RCA_F   3.5f
#define ETAR_F  16.0f
#define ETAA_F  8.0f
#define PI_F    3.14159265358979f
#define H2EV_F  27.211386024367243f

__device__ __constant__ float c_SHFR[16] = {0.9f,1.16875f,1.4375f,1.70625f,1.975f,2.24375f,2.5125f,2.78125f,3.05f,3.31875f,3.5875f,3.85625f,4.125f,4.39375f,4.6625f,4.93125f};
__device__ __constant__ float c_SHFZ[8]  = {0.19634954f,0.58904862f,0.9817477f,1.3744468f,1.7671459f,2.1598449f,2.552544f,2.9452431f};
__device__ __constant__ float c_SHFA[4]  = {0.9f,1.55f,2.2f,2.85f};
__device__ __constant__ float c_SAE[7]   = {-0.5984f,-38.0826f,-54.7031f,-75.1901f,-99.8006f,-398.1224f,-460.1387f};
__device__ __constant__ int   c_D1[7] = {256,224,192,192,160,160,160};
__device__ __constant__ int   c_D2[7] = {192,192,160,160,128,128,128};
__device__ __constant__ int   c_D3[7] = {160,160,128,128,96,96,96};

struct PackArgs {
    const float* Wsrc[NSPEC][3];
    _Float16*    Fdst[NSPEC][3];   // fwd: n-major, k-contiguous, K zero-padded
    _Float16*    Cdst[NSPEC][3];   // bwd: f16 cast of original row-major layout
};

struct MlpArgs {
    const _Float16* F[NSPEC][3];
    const _Float16* C[NSPEC][3];
    const float*    W3[NSPEC];
    const float*    Bias[NSPEC][4];
};

__device__ __forceinline__ int pair_index(int s1, int s2) {
    int a = min(s1, s2), b = max(s1, s2);
    return a * NSPEC - (a * (a - 1)) / 2 + (b - a);
}

// ------------------------------------------------------------------
// Weight pack kernel: f32 -> f16 cast (bwd) + transpose/pad (fwd).
// ------------------------------------------------------------------
__global__ __launch_bounds__(256) void ani_pack_kernel(PackArgs pk) {
    int s = blockIdx.y, l = blockIdx.z;
    int K    = (l == 0) ? AEV_DIM : (l == 1 ? c_D1[s] : c_D2[s]);
    int N    = (l == 0) ? c_D1[s] : (l == 1 ? c_D2[s] : c_D3[s]);
    int Kpad = (l == 0) ? AEV_PAD : K;
    const float* src = pk.Wsrc[s][l];
    _Float16* C = pk.Cdst[s][l];
    _Float16* F = pk.Fdst[s][l];
    int t0 = blockIdx.x * 256 + threadIdx.x;
    int stride = gridDim.x * 256;
    for (int idx = t0; idx < K * N; idx += stride)
        C[idx] = (_Float16)src[idx];
    for (int idx = t0; idx < N * Kpad; idx += stride) {
        int n = idx / Kpad, k = idx - n * Kpad;
        F[idx] = (k < K) ? (_Float16)src[(size_t)k * N + n] : (_Float16)0.f;
    }
}

// ------------------------------------------------------------------
// Kernel 1: AEV forward.  One wave32 per center atom.
// ------------------------------------------------------------------
__global__ __launch_bounds__(32) void ani_aev_kernel(const int* __restrict__ species,
                                                     const float* __restrict__ coords,
                                                     _Float16* __restrict__ aev_h) {
    int center = blockIdx.x;
    int b = center >> 5, i = center & 31;
    int tid = threadIdx.x;

    __shared__ float buf[AEV_DIM];
    __shared__ float cx[NATOM], cy[NATOM], cz[NATOM];
    __shared__ int   sp[NATOM];

    {
        const float* c = coords + (size_t)(b * NATOM + tid) * 3;
        cx[tid] = c[0]; cy[tid] = c[1]; cz[tid] = c[2];
        sp[tid] = species[b * NATOM + tid];
    }
    for (int idx = tid; idx < AEV_DIM; idx += 32) buf[idx] = 0.f;
    __syncthreads();

    float xi = cx[i], yi = cy[i], zi = cz[i];

    // ---- radial: one neighbor j per lane ----
    {
        int j = tid;
        if (j != i) {
            float vx = cx[j] - xi, vy = cy[j] - yi, vz = cz[j] - zi;
            float d = sqrtf(vx * vx + vy * vy + vz * vz + 1e-12f);
            if (d <= RCR_F) {
                float fc = 0.5f * __cosf(PI_F * d / RCR_F) + 0.5f;
                int base = sp[j] * 16;
#pragma unroll
                for (int t = 0; t < 16; ++t) {
                    float dd = d - c_SHFR[t];
                    atomicAdd(&buf[base + t], 0.25f * __expf(-ETAR_F * dd * dd) * fc);
                }
            }
        }
    }

    // ---- angular: unordered pairs (j,k), counted x2 ----
    for (int p = tid; p < NATOM * NATOM; p += 32) {
        int j = p >> 5, k = p & 31;
        if (j >= k || j == i || k == i) continue;
        float vjx = cx[j] - xi, vjy = cy[j] - yi, vjz = cz[j] - zi;
        float vkx = cx[k] - xi, vky = cy[k] - yi, vkz = cz[k] - zi;
        float dij = sqrtf(vjx * vjx + vjy * vjy + vjz * vjz + 1e-12f);
        float dik = sqrtf(vkx * vkx + vky * vky + vkz * vkz + 1e-12f);
        if (dij > RCA_F || dik > RCA_F) continue;
        float fcJ = 0.5f * __cosf(PI_F * dij / RCA_F) + 0.5f;
        float fcK = 0.5f * __cosf(PI_F * dik / RCA_F) + 0.5f;
        float ct = (vjx * vkx + vjy * vky + vjz * vkz) / (dij * dik);
        float u = 0.95f * fminf(1.f, fmaxf(-1.f, ct));
        float th = acosf(u);
        float ravg = 0.5f * (dij + dik);
        float fe[4];
#pragma unroll
        for (int a = 0; a < 4; ++a) {
            float rr = ravg - c_SHFA[a];
            fe[a] = __expf(-ETAA_F * rr * rr);
        }
        float fac = 2.f * fcJ * fcK;
        int base = NSPEC * 16 + pair_index(sp[j], sp[k]) * 32;
#pragma unroll
        for (int z = 0; z < 8; ++z) {
            float q = 0.5f * (1.f + __cosf(th - c_SHFZ[z]));
            float q2 = q * q, q4 = q2 * q2, q8 = q4 * q4, q16 = q8 * q8;
            float t0 = fac * q16 * q16;   // q^32
#pragma unroll
            for (int a = 0; a < 4; ++a)
                atomicAdd(&buf[base + z * 4 + a], t0 * fe[a]);
        }
    }
    __syncthreads();

    for (int idx = tid; idx < AEV_PAD; idx += 32)
        aev_h[(size_t)center * AEV_PAD + idx] = (_Float16)(idx < AEV_DIM ? buf[idx] : 0.f);
}

// ------------------------------------------------------------------
// WMMA fragment helpers.  ISA 7.12.2: a lane's 16 halves are two
// contiguous 8-half runs at k = khi and k = khi+16 -> 2x b128 loads.
// ------------------------------------------------------------------
__device__ __forceinline__ v16h load_frag16(const _Float16* p) {
    v8h lo = *(const v8h*)(p);
    v8h hi = *(const v8h*)(p + 16);
    return __builtin_shufflevector(lo, hi, 0,1,2,3,4,5,6,7,8,9,10,11,12,13,14,15);
}

// Z(32xN f32 LDS) = A(32xK f16 LDS, row-major lda) * B(KxN), where B is
// packed n-major/k-contiguous: element (k,n) at Wh[n*ldk + k].
__device__ void gemm_wmma(const _Float16* __restrict__ Am, int lda,
                          const _Float16* __restrict__ Wh, int ldk,
                          float* __restrict__ Zo, int ldz,
                          int N, int K, int tid) {
    int lane = tid & 31, wave = tid >> 5;
    int khi = (lane >= 16) ? 8 : 0;
    int total = (N >> 4) << 1;
    for (int t = wave; t < total; t += 4) {
        int mt = t & 1, nt = t >> 1;
        const _Float16* ap = Am + (size_t)(mt * 16 + (lane & 15)) * lda + khi;
        const _Float16* bp = Wh + (size_t)(nt * 16 + (lane & 15)) * ldk + khi;
        v8f acc = {0.f, 0.f, 0.f, 0.f, 0.f, 0.f, 0.f, 0.f};
        for (int k0 = 0; k0 < K; k0 += 32) {
            v16h a  = load_frag16(ap + k0);
            v16h bf = load_frag16(bp + k0);
            acc = __builtin_amdgcn_wmma_f32_16x16x32_f16(false, a, false, bf,
                                                         (short)0, acc, false, false);
        }
        int rbase = mt * 16 + khi;
        int col = nt * 16 + (lane & 15);
#pragma unroll
        for (int r = 0; r < 8; ++r)
            Zo[(size_t)(rbase + r) * ldz + col] = acc[r];
    }
}

// Final backward GEMM: dAEV rows = A(32xK) * W0^T, written straight to global.
__device__ void gemm_wmma_out(const _Float16* __restrict__ Am, int lda,
                              const _Float16* __restrict__ Wh, int ldk,
                              float* __restrict__ gaev,
                              const int* __restrict__ alist, int cnt, int atom0,
                              int N, int K, int tid) {
    int lane = tid & 31, wave = tid >> 5;
    int khi = (lane >= 16) ? 8 : 0;
    int total = (N >> 4) << 1;
    for (int t = wave; t < total; t += 4) {
        int mt = t & 1, nt = t >> 1;
        const _Float16* ap = Am + (size_t)(mt * 16 + (lane & 15)) * lda + khi;
        const _Float16* bp = Wh + (size_t)(nt * 16 + (lane & 15)) * ldk + khi;
        v8f acc = {0.f, 0.f, 0.f, 0.f, 0.f, 0.f, 0.f, 0.f};
        for (int k0 = 0; k0 < K; k0 += 32) {
            v16h a  = load_frag16(ap + k0);
            v16h bf = load_frag16(bp + k0);
            acc = __builtin_amdgcn_wmma_f32_16x16x32_f16(false, a, false, bf,
                                                         (short)0, acc, false, false);
        }
        int rbase = mt * 16 + khi;
        int col = nt * 16 + (lane & 15);
#pragma unroll
        for (int r = 0; r < 8; ++r) {
            int rr = rbase + r;
            if (rr < cnt)
                gaev[(size_t)(atom0 + alist[rr]) * AEV_PAD + col] = acc[r] * H2EV_F;
        }
    }
}

__device__ void epilogue_fwd(const float* __restrict__ Z, const float* __restrict__ bias,
                             _Float16* __restrict__ H, _Float16* __restrict__ Dm,
                             int N, int tid) {
    for (int idx = tid; idx < 32 * N; idx += 128) {
        int r = idx / N, n = idx - r * N;
        float z = Z[(size_t)r * 256 + n] + bias[n];
        float h = (z > 0.f) ? z : 0.1f * (__expf(z * 10.f) - 1.f);   // celu, alpha=0.1
        H[(size_t)r * 256 + n]  = (_Float16)h;
        Dm[(size_t)r * 256 + n] = (_Float16)((z > 0.f) ? 1.f : __expf(z * 10.f));
    }
}

__device__ void epilogue_bwd(const float* __restrict__ Z, const _Float16* __restrict__ Dm,
                             _Float16* __restrict__ dZ, int N, int tid) {
    for (int idx = tid; idx < 32 * N; idx += 128) {
        int r = idx / N, n = idx - r * N;
        dZ[(size_t)r * 256 + n] = (_Float16)(Z[(size_t)r * 256 + n] * (float)Dm[(size_t)r * 256 + n]);
    }
}

// ------------------------------------------------------------------
// Kernel 2: MLP forward + backward per (molecule, species) tile.
// ------------------------------------------------------------------
#define SM_X   (32 * AEV_PAD * 2)    // 65536 B  f16 X tile
#define SM_Z   (32 * 256 * 4)        // 32768 B  f32 GEMM output
#define SM_ACT (32 * 256 * 2)        // 16384 B  f16 buffers (x5)
#define SM_TOT (SM_X + SM_Z + 5 * SM_ACT + 33 * 4)

__global__ __launch_bounds__(128) void ani_mlp_kernel(const int* __restrict__ species,
                                                      const _Float16* __restrict__ aev_h,
                                                      float* __restrict__ gaev,
                                                      float* __restrict__ energy,
                                                      MlpArgs M) {
    int b = blockIdx.x, s = blockIdx.y;
    int tid = threadIdx.x;

    extern __shared__ char smem[];
    _Float16* X    = (_Float16*)smem;
    float*    Z    = (float*)(smem + SM_X);
    _Float16* actA = (_Float16*)(smem + SM_X + SM_Z);
    _Float16* actB = actA + 32 * 256;
    _Float16* D1m  = actB + 32 * 256;
    _Float16* D2m  = D1m + 32 * 256;
    _Float16* D3m  = D2m + 32 * 256;
    int*      alist = (int*)(smem + SM_X + SM_Z + 5 * SM_ACT);

    if (tid == 0) {
        int c = 0;
        for (int a = 0; a < NATOM; ++a)
            if (species[b * NATOM + a] == s) alist[c++] = a;
        alist[32] = c;
    }
    __syncthreads();
    int cnt = alist[32];
    if (cnt == 0) return;

    int d1 = c_D1[s], d2 = c_D2[s], d3 = c_D3[s];

    // load X tile with b128 copies (rows >= cnt zeroed)
    for (int idx = tid; idx < 32 * (AEV_PAD / 8); idx += 128) {
        int r = idx >> 7, cpos = (idx & 127) << 3;
        uint4 v = make_uint4(0u, 0u, 0u, 0u);
        if (r < cnt)
            v = *(const uint4*)(aev_h + (size_t)(b * NATOM + alist[r]) * AEV_PAD + cpos);
        *(uint4*)(X + (size_t)r * AEV_PAD + cpos) = v;
    }
    __syncthreads();

    // ---- forward ----
    gemm_wmma(X, AEV_PAD, M.F[s][0], AEV_PAD, Z, 256, d1, AEV_PAD, tid);
    __syncthreads();
    epilogue_fwd(Z, M.Bias[s][0], actA, D1m, d1, tid);                 // H1
    __syncthreads();
    gemm_wmma(actA, 256, M.F[s][1], d1, Z, 256, d2, d1, tid);
    __syncthreads();
    epilogue_fwd(Z, M.Bias[s][1], actB, D2m, d2, tid);                 // H2
    __syncthreads();
    gemm_wmma(actB, 256, M.F[s][2], d2, Z, 256, d3, d2, tid);
    __syncthreads();
    epilogue_fwd(Z, M.Bias[s][2], actA, D3m, d3, tid);                 // H3
    __syncthreads();

    // ---- output layer + energy ----
    if (tid < 32) {
        int r = tid;
        const float* W3 = M.W3[s];
        float acc = M.Bias[s][3][0];
        for (int k = 0; k < d3; ++k)
            acc += (float)actA[(size_t)r * 256 + k] * W3[k];
        if (r < cnt)
            atomicAdd(&energy[b], (acc + c_SAE[s]) * H2EV_F);
    }
    // dZ3 = W3 * celu'(z3)
    for (int idx = tid; idx < 32 * d3; idx += 128) {
        int r = idx / d3, k = idx - r * d3;
        actB[(size_t)r * 256 + k] =
            (_Float16)(M.W3[s][k] * (float)D3m[(size_t)r * 256 + k]);
    }
    __syncthreads();

    // ---- backward (B operands are f16 casts of original row-major W) ----
    gemm_wmma(actB, 256, M.C[s][2], d3, Z, 256, d2, d3, tid);          // dH2 = dZ3 @ W2^T
    __syncthreads();
    epilogue_bwd(Z, D2m, actA, d2, tid);                               // dZ2
    __syncthreads();
    gemm_wmma(actA, 256, M.C[s][1], d2, Z, 256, d1, d2, tid);          // dH1 = dZ2 @ W1^T
    __syncthreads();
    epilogue_bwd(Z, D1m, actB, d1, tid);                               // dZ1
    __syncthreads();
    gemm_wmma_out(actB, 256, M.C[s][0], d1, gaev, alist, cnt,
                  b * NATOM, AEV_DIM, d1, tid);                        // dAEV = dZ1 @ W0^T
}

// ------------------------------------------------------------------
// Kernel 3: AEV backward -> forces.  One wave32 per center atom.
// ------------------------------------------------------------------
__global__ __launch_bounds__(32) void ani_force_kernel(const int* __restrict__ species,
                                                       const float* __restrict__ coords,
                                                       const float* __restrict__ gaev,
                                                       float* __restrict__ force) {
    int center = blockIdx.x;
    int b = center >> 5, i = center & 31;
    int tid = threadIdx.x;

    __shared__ float g[AEV_DIM];
    __shared__ float cx[NATOM], cy[NATOM], cz[NATOM];
    __shared__ int   sp[NATOM];
    {
        const float* c = coords + (size_t)(b * NATOM + tid) * 3;
        cx[tid] = c[0]; cy[tid] = c[1]; cz[tid] = c[2];
        sp[tid] = species[b * NATOM + tid];
    }
    for (int idx = tid; idx < AEV_DIM; idx += 32)
        g[idx] = gaev[(size_t)center * AEV_PAD + idx];
    __syncthreads();

    float xi = cx[i], yi = cy[i], zi = cz[i];
    float fix = 0.f, fiy = 0.f, fiz = 0.f;

    // ---- radial backward ----
    {
        int j = tid;
        if (j != i) {
            float vx = cx[j] - xi, vy = cy[j] - yi, vz = cz[j] - zi;
            float d = sqrtf(vx * vx + vy * vy + vz * vz + 1e-12f);
            if (d <= RCR_F) {
                float fc  = 0.5f * __cosf(PI_F * d / RCR_F) + 0.5f;
                float fcp = -0.5f * (PI_F / RCR_F) * __sinf(PI_F * d / RCR_F);
                int base = sp[j] * 16;
                float S = 0.f;
#pragma unroll
                for (int t = 0; t < 16; ++t) {
                    float dd = d - c_SHFR[t];
                    float e = 0.25f * __expf(-ETAR_F * dd * dd);
                    S += g[base + t] * (e * (-2.f * ETAR_F * dd) * fc + e * fcp);
                }
                float inv = 1.f / d;
                float gx = S * vx * inv, gy = S * vy * inv, gz = S * vz * inv;
                atomicAdd(&force[(size_t)(b * NATOM + j) * 3 + 0], -gx);
                atomicAdd(&force[(size_t)(b * NATOM + j) * 3 + 1], -gy);
                atomicAdd(&force[(size_t)(b * NATOM + j) * 3 + 2], -gz);
                fix += gx; fiy += gy; fiz += gz;
            }
        }
    }

    // ---- angular backward ----
    for (int p = tid; p < NATOM * NATOM; p += 32) {
        int j = p >> 5, k = p & 31;
        if (j >= k || j == i || k == i) continue;
        float vjx = cx[j] - xi, vjy = cy[j] - yi, vjz = cz[j] - zi;
        float vkx = cx[k] - xi, vky = cy[k] - yi, vkz = cz[k] - zi;
        float dij = sqrtf(vjx * vjx + vjy * vjy + vjz * vjz + 1e-12f);
        float dik = sqrtf(vkx * vkx + vky * vky + vkz * vkz + 1e-12f);
        if (dij > RCA_F || dik > RCA_F) continue;
        float fcJ  = 0.5f * __cosf(PI_F * dij / RCA_F) + 0.5f;
        float fcK  = 0.5f * __cosf(PI_F * dik / RCA_F) + 0.5f;
        float fcpJ = -0.5f * (PI_F / RCA_F) * __sinf(PI_F * dij / RCA_F);
        float fcpK = -0.5f * (PI_F / RCA_F) * __sinf(PI_F * dik / RCA_F);
        float dotv = vjx * vkx + vjy * vky + vjz * vkz;
        float ct = dotv / (dij * dik);
        float u = 0.95f * fminf(1.f, fmaxf(-1.f, ct));
        float th = acosf(u);
        float dthdc = (fabsf(ct) < 1.f) ? (-0.95f * rsqrtf(1.f - u * u)) : 0.f;
        float ravg = 0.5f * (dij + dik);
        float fe[4], dfe[4];
#pragma unroll
        for (int a = 0; a < 4; ++a) {
            float rr = ravg - c_SHFA[a];
            fe[a] = __expf(-ETAA_F * rr * rr);
            dfe[a] = fe[a] * (-2.f * ETAA_F * rr);
        }
        int base = NSPEC * 16 + pair_index(sp[j], sp[k]) * 32;
        float S0 = 0.f, S1 = 0.f, S2 = 0.f;
#pragma unroll
        for (int z = 0; z < 8; ++z) {
            float ang = th - c_SHFZ[z];
            float q = 0.5f * (1.f + __cosf(ang));
            float q2 = q * q, q4 = q2 * q2, q8 = q4 * q4, q16 = q8 * q8;
            float q31 = q16 * q8 * q4 * q2 * q;
            float fz = q31 * q;
            float dfz = -16.f * __sinf(ang) * q31;
#pragma unroll
            for (int a = 0; a < 4; ++a) {
                float gv = g[base + z * 4 + a];
                S0 += gv * fz * fe[a];
                S1 += gv * dfz * fe[a];
                S2 += gv * fz * dfe[a];
            }
        }
        float F = fcJ * fcK;
        float dEdth   = 2.f * S1 * F;
        float dEdravg = 2.f * S2 * F;
        float dEddij  = 0.5f * dEdravg + 2.f * S0 * fcpJ * fcK;
        float dEddik  = 0.5f * dEdravg + 2.f * S0 * fcJ * fcpK;
        float invij = 1.f / dij, invik = 1.f / dik, invjk = invij * invik;
        float A1 = dEdth * dthdc;
        float gjx = A1 * (vkx * invjk - ct * vjx * invij * invij) + dEddij * vjx * invij;
        float gjy = A1 * (vky * invjk - ct * vjy * invij * invij) + dEddij * vjy * invij;
        float gjz = A1 * (vkz * invjk - ct * vjz * invij * invij) + dEddij * vjz * invij;
        float gkx = A1 * (vjx * invjk - ct * vkx * invik * invik) + dEddik * vkx * invik;
        float gky = A1 * (vjy * invjk - ct * vky * invik * invik) + dEddik * vky * invik;
        float gkz = A1 * (vjz * invjk - ct * vkz * invik * invik) + dEddik * vkz * invik;
        atomicAdd(&force[(size_t)(b * NATOM + j) * 3 + 0], -gjx);
        atomicAdd(&force[(size_t)(b * NATOM + j) * 3 + 1], -gjy);
        atomicAdd(&force[(size_t)(b * NATOM + j) * 3 + 2], -gjz);
        atomicAdd(&force[(size_t)(b * NATOM + k) * 3 + 0], -gkx);
        atomicAdd(&force[(size_t)(b * NATOM + k) * 3 + 1], -gky);
        atomicAdd(&force[(size_t)(b * NATOM + k) * 3 + 2], -gkz);
        fix += gjx + gkx; fiy += gjy + gky; fiz += gjz + gkz;
    }

    atomicAdd(&force[(size_t)(b * NATOM + i) * 3 + 0], fix);
    atomicAdd(&force[(size_t)(b * NATOM + i) * 3 + 1], fiy);
    atomicAdd(&force[(size_t)(b * NATOM + i) * 3 + 2], fiz);
}

// ------------------------------------------------------------------
extern "C" void kernel_launch(void* const* d_in, const int* in_sizes, int n_in,
                              void* d_out, int out_size, void* d_ws, size_t ws_size,
                              hipStream_t stream) {
    (void)in_sizes; (void)n_in; (void)ws_size;
    const int*   species = (const int*)d_in[0];
    const float* coords  = (const float*)d_in[1];

    static const int D1[7] = {256,224,192,192,160,160,160};
    static const int D2[7] = {192,192,160,160,128,128,128};
    static const int D3[7] = {160,160,128,128,96,96,96};

    _Float16* aev_h = (_Float16*)d_ws;                                     // 2048*1024 f16
    float*    gaev  = (float*)((char*)d_ws + (size_t)2048 * AEV_PAD * 2);  // 2048*1024 f32
    _Float16* wp    = (_Float16*)((char*)gaev + (size_t)2048 * AEV_PAD * 4);

    PackArgs pk;
    MlpArgs  M;
    for (int s = 0; s < NSPEC; ++s) {
        for (int l = 0; l < 4; ++l)
            M.Bias[s][l] = (const float*)d_in[2 + 8 * s + 4 + l];
        M.W3[s] = (const float*)d_in[2 + 8 * s + 3];
        int d1 = D1[s], d2 = D2[s], d3 = D3[s];
        int Kp[3] = {AEV_PAD, d1, d2};
        int Ks[3] = {AEV_DIM, d1, d2};
        int Ns[3] = {d1, d2, d3};
        for (int l = 0; l < 3; ++l) {
            pk.Wsrc[s][l] = (const float*)d_in[2 + 8 * s + l];
            pk.Fdst[s][l] = wp;  M.F[s][l] = wp;  wp += (size_t)Ns[l] * Kp[l];
            pk.Cdst[s][l] = wp;  M.C[s][l] = wp;  wp += (size_t)Ks[l] * Ns[l];
        }
    }

    float* energy = (float*)d_out;
    float* force  = energy + NMOL;

    hipMemsetAsync(d_out, 0, (size_t)out_size * sizeof(float), stream);

    ani_pack_kernel<<<dim3(16, NSPEC, 3), 256, 0, stream>>>(pk);
    ani_aev_kernel<<<NMOL * NATOM, 32, 0, stream>>>(species, coords, aev_h);
    ani_mlp_kernel<<<dim3(NMOL, NSPEC), 128, SM_TOT, stream>>>(species, aev_h, gaev, energy, M);
    ani_force_kernel<<<NMOL * NATOM, 32, 0, stream>>>(species, coords, gaev, force);
}
